// UpsampleBlock_8598524526863
// MI455X (gfx1250) — matrix operand order repaired and checked by reading, not compiled, not run on hardware
//
#include <hip/hip_runtime.h>
#include <cstdint>
#include <cstddef>

typedef __attribute__((ext_vector_type(16))) _Float16 v16h;
typedef __attribute__((ext_vector_type(8)))  _Float16 v8h;
typedef __attribute__((ext_vector_type(8)))  float    v8f;
typedef __attribute__((ext_vector_type(4)))  int      v4i;
typedef __attribute__((ext_vector_type(4)))  unsigned int uint32x4;
typedef __attribute__((ext_vector_type(8)))  int      int32x8;
typedef __attribute__((ext_vector_type(4)))  int      int32x4;

#define BN_EPS 1e-5f

#if __has_builtin(__builtin_amdgcn_tensor_load_to_lds) && \
    __has_builtin(__builtin_amdgcn_s_wait_tensorcnt)
#define HAVE_TDM 1
#endif
#if __has_builtin(__builtin_amdgcn_global_load_async_to_lds_b128)
#define HAVE_ASYNC 1
#endif

constexpr int B = 8, CIN = 64, COUT = 32;
constexpr int ATT = 16;
constexpr int KT  = 18;                      // 576 / 32 K-chunks (r-major: k = r*64 + ci)
constexpr int CIP = 72;                      // LDS ci pitch in halfs (16B aligned fragments)
constexpr int COLS = 130;                    // halo columns ow0-1 .. ow0+128
constexpr int LDS_B_BYTES = 3 * COLS * CIP * 2;   // 56160
constexpr int LDS_A_BYTES = 2 * KT * 32 * 32;     // 36864 : [m][t][chunk][lane][16B]
constexpr int LDS_TOTAL   = LDS_B_BYTES + LDS_A_BYTES;

// workspace layout (bytes)
constexpr size_t WS_G    = 0;         // B*CIN f32   GAP
constexpr size_t WS_CH   = 4096;      // B*CIN f32   channel attention
constexpr size_t WS_FIL  = 8192;      // B*COUT f32  filter attention
constexpr size_t WS_SP   = 12288;     // B*9 f32     spatial attention
constexpr size_t WS_KATT = 13312;     // B*2 f32     kernel softmax
constexpr size_t WS_BNS  = 13824;     // COUT f32    bn scale
constexpr size_t WS_BIAS = 14336;     // COUT f32    bn bias
constexpr size_t WS_APK  = 16384;     // B*36864 B   packed f16 A fragments
constexpr size_t WS_XUP  = 524288;    // B*256*256*64 f16 channel-last upsampled x (64 MiB)

// ---------------------------------------------------------------------------
// Kernel 1: bilinear x2 (align_corners) upsample, fp32 NCHW -> f16 NHWC
// ---------------------------------------------------------------------------
__global__ void __launch_bounds__(256) upsample_kernel(
    const float* __restrict__ x, _Float16* __restrict__ xup)
{
  __shared__ float ls[2][64][34];   // [row][ci][iw_local]
  const int tid = threadIdx.x;
  const int b = blockIdx.z, oh = blockIdx.y, ow0 = blockIdx.x * 64;
  const float c = 127.0f / 255.0f;

  float posh = (float)oh * c;
  int ih0 = (int)posh;
  float fh = posh - (float)ih0;
  int ih1 = min(ih0 + 1, 127);
  int iwbase = (int)((float)ow0 * c);
  int rows2[2] = {ih0, ih1};

  for (int i = tid; i < 64 * 2 * 34; i += 256) {
    int ci = i / 68, rem = i - ci * 68;
    int rrow = rem / 34, iwl = rem - rrow * 34;
    int iw = min(iwbase + iwl, 127);
    ls[rrow][ci][iwl] = x[(((size_t)b * 64 + ci) * 128 + rows2[rrow]) * 128 + iw];
  }
  __syncthreads();

  const int ci = tid & 63;
  const int grp = tid >> 6;      // 0..3
  #pragma unroll
  for (int k = 0; k < 16; ++k) {
    int ow = ow0 + grp * 16 + k;
    float posw = (float)ow * c;
    int iw0 = (int)posw;
    float fw = posw - (float)iw0;
    int il0 = iw0 - iwbase;
    int il1 = min(iw0 + 1, 127) - iwbase;
    float v0 = ls[0][ci][il0] * (1.0f - fw) + ls[0][ci][il1] * fw;
    float v1 = ls[1][ci][il0] * (1.0f - fw) + ls[1][ci][il1] * fw;
    float v  = v0 * (1.0f - fh) + v1 * fh;
    xup[((size_t)b * 65536 + (size_t)oh * 256 + ow) * 64 + ci] = (_Float16)v;
  }
}

// ---------------------------------------------------------------------------
// Kernel 2: exact GAP of upsampled image via separable interp-weight sums
// ---------------------------------------------------------------------------
__global__ void __launch_bounds__(256) gap_kernel(
    const float* __restrict__ x, float* __restrict__ g)
{
  __shared__ float wrow[128];
  __shared__ float red[256];
  const int tid = threadIdx.x;
  const int ci = blockIdx.x, b = blockIdx.y;

  if (tid < 128) wrow[tid] = 0.0f;
  __syncthreads();
  {
    float pos = (float)tid * (127.0f / 255.0f);
    int i0 = (int)pos;
    float f = pos - (float)i0;
    int i1 = min(i0 + 1, 127);
    atomicAdd(&wrow[i0], 1.0f - f);
    atomicAdd(&wrow[i1], f);
  }
  __syncthreads();

  float sum = 0.0f;
  const float* xp = x + ((size_t)b * 64 + ci) * 16384;
  for (int i = tid; i < 16384; i += 256) {
    int r = i >> 7, cc = i & 127;
    sum += wrow[r] * wrow[cc] * xp[i];
  }
  red[tid] = sum;
  __syncthreads();
  for (int s = 128; s > 0; s >>= 1) {
    if (tid < s) red[tid] += red[tid + s];
    __syncthreads();
  }
  if (tid == 0) g[b * 64 + ci] = red[0] * (1.0f / 65536.0f);
}

// ---------------------------------------------------------------------------
// Kernel 3: attention MLP heads (tiny; one block)
// ---------------------------------------------------------------------------
__device__ __forceinline__ float sigmoidf_(float x) { return 1.0f / (1.0f + __expf(-x)); }

__global__ void __launch_bounds__(256) attn_kernel(
    const float* __restrict__ g,
    const float* __restrict__ fc_w,
    const float* __restrict__ bna_g, const float* __restrict__ bna_b,
    const float* __restrict__ bna_m, const float* __restrict__ bna_v,
    const float* __restrict__ ch_w, const float* __restrict__ ch_b,
    const float* __restrict__ fil_w, const float* __restrict__ fil_b,
    const float* __restrict__ sp_w, const float* __restrict__ sp_b,
    const float* __restrict__ k_w, const float* __restrict__ k_b,
    const float* __restrict__ bn_g, const float* __restrict__ bn_b,
    const float* __restrict__ bn_m, const float* __restrict__ bn_v,
    float* __restrict__ ws_ch, float* __restrict__ ws_fil,
    float* __restrict__ ws_sp, float* __restrict__ ws_k,
    float* __restrict__ ws_bns, float* __restrict__ ws_bias)
{
  __shared__ float hl[B * ATT];
  __shared__ float klog[B * 2];
  const int tid = threadIdx.x;

  if (tid < B * ATT) {
    int b = tid >> 4, a = tid & 15;
    float s = 0.0f;
    for (int ci = 0; ci < CIN; ++ci) s += g[b * CIN + ci] * fc_w[a * CIN + ci];
    s = (s - bna_m[a]) * rsqrtf(bna_v[a] + BN_EPS) * bna_g[a] + bna_b[a];
    hl[tid] = fmaxf(s, 0.0f);
  }
  __syncthreads();

  for (int i = tid; i < B * CIN; i += 256) {       // channel attention
    int b = i >> 6, ci = i & 63;
    float s = ch_b[ci];
    for (int a = 0; a < ATT; ++a) s += hl[b * ATT + a] * ch_w[ci * ATT + a];
    ws_ch[i] = sigmoidf_(s);
  }
  if (tid < B * COUT) {                            // filter attention
    int b = tid >> 5, co = tid & 31;
    float s = fil_b[co];
    for (int a = 0; a < ATT; ++a) s += hl[b * ATT + a] * fil_w[co * ATT + a];
    ws_fil[tid] = sigmoidf_(s);
  }
  if (tid < B * 9) {                               // spatial attention
    int b = tid / 9, r = tid - b * 9;
    float s = sp_b[r];
    for (int a = 0; a < ATT; ++a) s += hl[b * ATT + a] * sp_w[r * ATT + a];
    ws_sp[tid] = sigmoidf_(s);
  }
  if (tid < B * 2) {                               // kernel logits
    int b = tid >> 1, k = tid & 1;
    float s = k_b[k];
    for (int a = 0; a < ATT; ++a) s += hl[b * ATT + a] * k_w[k * ATT + a];
    klog[tid] = s;
  }
  __syncthreads();
  if (tid < B) {                                   // softmax over K=2
    float m = fmaxf(klog[2 * tid], klog[2 * tid + 1]);
    float e0 = __expf(klog[2 * tid] - m), e1 = __expf(klog[2 * tid + 1] - m);
    float inv = 1.0f / (e0 + e1);
    ws_k[2 * tid] = e0 * inv;
    ws_k[2 * tid + 1] = e1 * inv;
  }
  if (tid < COUT) {                                // bn scale / bias
    float s = bn_g[tid] * rsqrtf(bn_v[tid] + BN_EPS);
    ws_bns[tid] = s;
    ws_bias[tid] = bn_b[tid] - bn_m[tid] * s;
  }
}

// ---------------------------------------------------------------------------
// Kernel 4: fold all attentions + bn scale into per-sample f16 A fragments,
// packed exactly in the gfx1250 16-bit A-matrix 16x32 WMMA layout.
// apack byte layout: [b][m][t][chunk(2)][lane(32)][16B]
// ---------------------------------------------------------------------------
__global__ void __launch_bounds__(256) pack_kernel(
    const float* __restrict__ weight,
    const float* __restrict__ katt, const float* __restrict__ sp,
    const float* __restrict__ ch, const float* __restrict__ fil,
    const float* __restrict__ bns, _Float16* __restrict__ apack)
{
  int gid = blockIdx.x * 256 + threadIdx.x;
  if (gid >= B * 2 * KT * 32) return;
  int lane = gid & 31;
  int frag = gid >> 5;
  int t = frag % KT;
  int m = (frag / KT) & 1;
  int b = frag / (KT * 2);
  int r = t >> 1, ci0 = (t & 1) << 5;
  int co = m * 16 + (lane & 15);
  int basek = (lane >> 4) << 3;     // lanes 0-15: K{0-7,16-23}; lanes 16-31: K{8-15,24-31}

  float ka0 = katt[b * 2], ka1 = katt[b * 2 + 1];
  float sc = sp[b * 9 + r] * fil[b * COUT + co] * bns[co];

  alignas(16) _Float16 hb[16];
  #pragma unroll
  for (int j = 0; j < 16; ++j) {
    int klocal = (j < 8) ? (basek + j) : (basek + 8 + j);   // j>=8 -> basek+16+(j-8)
    int ci = ci0 + klocal;
    float w0 = weight[((size_t)(0 * COUT + co) * CIN + ci) * 9 + r];
    float w1 = weight[((size_t)(1 * COUT + co) * CIN + ci) * 9 + r];
    float v = (ka0 * w0 + ka1 * w1) * sc * ch[b * CIN + ci];
    hb[j] = (_Float16)v;
  }
  char* dst = (char*)apack + (size_t)b * LDS_A_BYTES + (size_t)(m * KT + t) * 1024 + lane * 16;
  *(uint4*)dst         = *(const uint4*)&hb[0];   // chunk0 (VGPR0-3)
  *(uint4*)(dst + 512) = *(const uint4*)&hb[8];   // chunk1 (VGPR4-7)
}

// ---------------------------------------------------------------------------
// Kernel 5: implicit-GEMM 3x3 conv with v_wmma_f32_16x16x32_f16, TDM-staged
// weights, async global->LDS halo staging, fused bias/GELU epilogue.
// ---------------------------------------------------------------------------
__global__ void __launch_bounds__(256) conv_wmma_kernel(
    const _Float16* __restrict__ xup, const _Float16* __restrict__ apack,
    const float* __restrict__ bias, float* __restrict__ out)
{
  extern __shared__ char lds[];
  char* ldsB = lds;
  char* ldsA = lds + LDS_B_BYTES;
  const int tid = threadIdx.x;
  const int b = blockIdx.z, oh = blockIdx.y, ow0 = blockIdx.x * 128;

  // --- stage 3x130x64 f16 halo (channel-last, ci pitch 72 halfs), zero pad ---
  for (int i = tid; i < 3 * COLS * 8; i += 256) {
    int pair = i >> 3, ci8 = (i & 7) << 3;
    int dh = pair / COLS, w = pair - dh * COLS;
    int hh = oh - 1 + dh, ww = ow0 - 1 + w;
    char* lp = ldsB + ((size_t)(dh * COLS + w) * CIP + ci8) * 2;
    if ((unsigned)hh < 256u && (unsigned)ww < 256u) {
      const _Float16* gp = xup + (((size_t)b * 65536 + (size_t)hh * 256 + ww) * 64 + ci8);
#if defined(HAVE_ASYNC)
      __builtin_amdgcn_global_load_async_to_lds_b128(
          (__attribute__((address_space(1))) v4i*)(uintptr_t)gp,
          (__attribute__((address_space(3))) v4i*)(uintptr_t)lp, 0, 0);
#else
      *(uint4*)lp = *(const uint4*)gp;
#endif
    } else {
      uint4 z = {0u, 0u, 0u, 0u};
      *(uint4*)lp = z;
    }
  }

  // --- stage this sample's packed A (36864 B contiguous) via TDM ---
#if defined(HAVE_TDM)
  if (tid < 32) {  // one wave issues the tensor DMA descriptor
    const uint64_t ga = (uint64_t)(uintptr_t)(apack + (size_t)b * (LDS_A_BYTES / 2));
    const uint32_t la = (uint32_t)(uintptr_t)ldsA;
    const unsigned NE = (unsigned)(LDS_A_BYTES / 2);   // 18432 2-byte elements
    uint32x4 g0;
    g0[0] = 1u;                                        // count=1 (valid), load, user
    g0[1] = la;                                        // lds_addr
    g0[2] = (unsigned)ga;                              // global_addr[31:0]
    g0[3] = (unsigned)((ga >> 32) & 0x01FFFFFFu) | 0x80000000u;  // addr[56:32] | type=2
    int32x8 g1;
    g1[0] = (int)(1u << 16);                           // data_size=2B; no pad/iterate
    g1[1] = (int)((NE & 0xFFFFu) << 16);               // tensor_dim0[15:0]
    g1[2] = (int)((NE >> 16) | (1u << 16));            // tensor_dim0[31:16]; tensor_dim1=1
    g1[3] = (int)((NE & 0xFFFFu) << 16);               // tile_dim0 = 18432
    g1[4] = 0;                                         // tile_dim1/2 unused (1-D tile)
    g1[5] = (int)NE;                                   // tensor_dim0_stride lo
    g1[6] = 0; g1[7] = 0;
    int32x4 g2 = {0, 0, 0, 0};
    int32x4 g3 = {0, 0, 0, 0};
#if __clang_major__ >= 23
    int32x8 g4 = {0, 0, 0, 0, 0, 0, 0, 0};
    __builtin_amdgcn_tensor_load_to_lds(g0, g1, g2, g3, g4, 0);
#else
    __builtin_amdgcn_tensor_load_to_lds(g0, g1, g2, g3, 0);
#endif
    __builtin_amdgcn_s_wait_tensorcnt(0);
  }
#else
  {
    const uint4* ap = (const uint4*)(apack + (size_t)b * (LDS_A_BYTES / 2));
    uint4* al = (uint4*)ldsA;
    for (int i = tid; i < LDS_A_BYTES / 16; i += 256) al[i] = ap[i];
  }
#endif

#if defined(HAVE_ASYNC)
#if __has_builtin(__builtin_amdgcn_s_wait_asynccnt)
  __builtin_amdgcn_s_wait_asynccnt(0);
#else
  asm volatile("s_wait_asynccnt 0x0" ::: "memory");
#endif
#endif
  __syncthreads();

  const int lane = tid & 31;
  const int wv = tid >> 5;
  const int nloc = (wv << 4) + (lane & 15);    // pixel within 128-wide tile
  const int cih = (lane >> 4) << 3;            // B frag K base: 0 | 8

  union V16 { v16h v; v8h h[2]; };
  v8f acc0 = {}; v8f acc1 = {};

  #pragma unroll
  for (int t = 0; t < KT; ++t) {
    const int r = t >> 1, ci0 = (t & 1) << 5;
    const int dh = r / 3, dw = r - dh * 3;
    // B fragment: lane holds N=lane%16, K = ci0 + cih + {0..7, 16..23}
    const char* bp = ldsB + ((size_t)((dh * COLS + (nloc + dw)) * CIP) + ci0 + cih) * 2;
    V16 bf; bf.h[0] = *(const v8h*)bp; bf.h[1] = *(const v8h*)(bp + 32);
    // A fragments (prepacked in HW layout): two b128 per lane each
    const char* a0p = ldsA + (size_t)t * 1024 + lane * 16;
    const char* a1p = a0p + KT * 1024;
    V16 a0; a0.h[0] = *(const v8h*)a0p; a0.h[1] = *(const v8h*)(a0p + 512);
    V16 a1; a1.h[0] = *(const v8h*)a1p; a1.h[1] = *(const v8h*)(a1p + 512);
    acc0 = __builtin_amdgcn_wmma_f32_16x16x32_f16(false, a0.v, false, bf.v,
                                                  (short)0, acc0, false, false);
    acc1 = __builtin_amdgcn_wmma_f32_16x16x32_f16(false, a1.v, false, bf.v,
                                                  (short)0, acc1, false, false);
  }

  // epilogue: C/D layout -> VGPR rr: lanes0-15 M=rr, lanes16-31 M=rr+8
  const int cobase = (lane >> 4) << 3;
  const int ow = ow0 + nloc;
  const size_t pix = (size_t)oh * 256 + ow;
  #pragma unroll
  for (int rr = 0; rr < 8; ++rr) {
    int coA = rr + cobase;
    int coB = coA + 16;
    float vA = acc0[rr] + bias[coA];
    float vB = acc1[rr] + bias[coB];
    vA = 0.5f * vA * (1.0f + erff(vA * 0.70710678118654752f));
    vB = 0.5f * vB * (1.0f + erff(vB * 0.70710678118654752f));
    out[((size_t)b * COUT + coA) * 65536 + pix] = vA;
    out[((size_t)b * COUT + coB) * 65536 + pix] = vB;
  }
}

// ---------------------------------------------------------------------------
extern "C" void kernel_launch(void* const* d_in, const int* in_sizes, int n_in,
                              void* d_out, int out_size, void* d_ws, size_t ws_size,
                              hipStream_t stream) {
  (void)in_sizes; (void)n_in; (void)out_size; (void)ws_size;
  const float* x      = (const float*)d_in[0];
  const float* fc_w   = (const float*)d_in[1];
  const float* bna_g  = (const float*)d_in[2];
  const float* bna_b  = (const float*)d_in[3];
  const float* bna_m  = (const float*)d_in[4];
  const float* bna_v  = (const float*)d_in[5];
  const float* ch_w   = (const float*)d_in[6];
  const float* ch_b   = (const float*)d_in[7];
  const float* fil_w  = (const float*)d_in[8];
  const float* fil_b  = (const float*)d_in[9];
  const float* sp_w   = (const float*)d_in[10];
  const float* sp_b   = (const float*)d_in[11];
  const float* k_w    = (const float*)d_in[12];
  const float* k_b    = (const float*)d_in[13];
  const float* weight = (const float*)d_in[14];
  const float* bn_g   = (const float*)d_in[15];
  const float* bn_b   = (const float*)d_in[16];
  const float* bn_m   = (const float*)d_in[17];
  const float* bn_v   = (const float*)d_in[18];
  float* out = (float*)d_out;

  char* ws = (char*)d_ws;
  float* ws_g    = (float*)(ws + WS_G);
  float* ws_ch   = (float*)(ws + WS_CH);
  float* ws_fil  = (float*)(ws + WS_FIL);
  float* ws_sp   = (float*)(ws + WS_SP);
  float* ws_k    = (float*)(ws + WS_KATT);
  float* ws_bns  = (float*)(ws + WS_BNS);
  float* ws_bias = (float*)(ws + WS_BIAS);
  _Float16* apack = (_Float16*)(ws + WS_APK);
  _Float16* xup   = (_Float16*)(ws + WS_XUP);

  upsample_kernel<<<dim3(4, 256, 8), 256, 0, stream>>>(x, xup);
  gap_kernel<<<dim3(64, 8), 256, 0, stream>>>(x, ws_g);
  attn_kernel<<<1, 256, 0, stream>>>(ws_g, fc_w, bna_g, bna_b, bna_m, bna_v,
                                     ch_w, ch_b, fil_w, fil_b, sp_w, sp_b, k_w, k_b,
                                     bn_g, bn_b, bn_m, bn_v,
                                     ws_ch, ws_fil, ws_sp, ws_k, ws_bns, ws_bias);
  pack_kernel<<<36, 256, 0, stream>>>(weight, ws_k, ws_sp, ws_ch, ws_fil, ws_bns, apack);

  static bool attr_done = false;
  if (!attr_done) {
    (void)hipFuncSetAttribute(reinterpret_cast<const void*>(conv_wmma_kernel),
                              hipFuncAttributeMaxDynamicSharedMemorySize, LDS_TOTAL);
    attr_done = true;
  }
  conv_wmma_kernel<<<dim3(2, 256, 8), 256, LDS_TOTAL, stream>>>(xup, apack, ws_bias, out);
}
